// WavLMEncoderLayer_38955353375362
// MI455X (gfx1250) — matrix operand order repaired and checked by reading, not compiled, not run on hardware
//
#include <hip/hip_runtime.h>
#include <math.h>

// ---------------- problem constants ----------------
constexpr int Bv = 2, Sv = 1500, Cv = 1024, Hv = 16, Dv = 64, Fv = 4096;

// ---------------- types & helpers ----------------
typedef __attribute__((ext_vector_type(16))) __bf16 bf16x16;
typedef __attribute__((ext_vector_type(8)))  float  f32x8;
typedef __attribute__((ext_vector_type(4)))  int    v4i;
typedef __attribute__((address_space(1))) v4i as1_v4i;   // global v4i
typedef __attribute__((address_space(3))) v4i as3_v4i;   // LDS v4i

union Frag { bf16x16 v; unsigned u[8]; };

__device__ __forceinline__ unsigned short f2bf(float f) {
  unsigned u = __builtin_bit_cast(unsigned, f);
  u += 0x7FFFu + ((u >> 16) & 1u);           // round-to-nearest-even
  return (unsigned short)(u >> 16);
}
__device__ __forceinline__ float bf2f(unsigned short h) {
  unsigned u = ((unsigned)h) << 16;
  return __builtin_bit_cast(float, u);
}
// A-matrix 16x32 bf16 K-pair base for vgpr j, lane-half (ISA 7.12.2)
__device__ __forceinline__ int kA(int j, int half) {
  return (j < 4) ? (2 * j + 8 * half) : (16 + 2 * (j - 4) + 8 * half);
}
// B-matrix 32x16 bf16 K-pair base for vgpr j, lane-half (same order as A)
__device__ __forceinline__ int kB(int j, int half) {
  return (j < 4) ? (2 * j + 8 * half) : (16 + 2 * (j - 4) + 8 * half);
}
__device__ __forceinline__ f32x8 wmma_bf16(const Frag& a, const Frag& b, f32x8 c) {
  return __builtin_amdgcn_wmma_f32_16x16x32_bf16(false, a.v, false, b.v,
                                                 (short)0, c, false, false);
}

// ---- gfx1250 async global->LDS copy (16B per lane), with portable fallback
__device__ __forceinline__ void async_copy16(const void* g, void* l) {
#if __has_builtin(__builtin_amdgcn_global_load_async_to_lds_b128)
  __builtin_amdgcn_global_load_async_to_lds_b128((as1_v4i*)g, (as3_v4i*)l, 0, 0);
#else
  *(uint4*)l = *(const uint4*)g;
#endif
}
__device__ __forceinline__ void wait_async() {
#if __has_builtin(__builtin_amdgcn_s_wait_asynccnt)
  __builtin_amdgcn_s_wait_asynccnt(0);
#else
  __asm volatile("s_wait_asynccnt 0" ::: "memory");
#endif
}

__device__ __forceinline__ float block_sum256(float v, float* sbuf) {
  #pragma unroll
  for (int off = 16; off > 0; off >>= 1) v += __shfl_xor(v, off, 32);
  int w = threadIdx.x >> 5;
  if ((threadIdx.x & 31) == 0) sbuf[w] = v;
  __syncthreads();
  float tot = 0.f;
  #pragma unroll
  for (int i = 0; i < 8; ++i) tot += sbuf[i];
  __syncthreads();
  return tot;
}

// ---------------- elementwise converts ----------------
__global__ __launch_bounds__(256) void cvt_f32_bf16_kernel(
    const float* __restrict__ in, unsigned short* __restrict__ out, int n) {
  int i = blockIdx.x * 256 + threadIdx.x;
  if (i < n) out[i] = f2bf(in[i]);
}

// x (B,C,S) f32 -> xt (B,S,C) bf16
__global__ __launch_bounds__(256) void xpose_kernel(
    const float* __restrict__ x, unsigned short* __restrict__ xt) {
  int i = blockIdx.x * 256 + threadIdx.x;
  if (i >= Bv * Sv * Cv) return;
  int c = i % Cv;
  int s = (i / Cv) % Sv;
  int b = i / (Cv * Sv);
  xt[i] = f2bf(x[((size_t)b * Cv + c) * Sv + s]);
}

// ---------------- gate: ga*(gb*gru-1)+2 per (b,h,s) ----------------
__global__ __launch_bounds__(256) void gate_kernel(
    const unsigned short* __restrict__ xt16, const float* __restrict__ Wg,
    const float* __restrict__ bg, const float* __restrict__ gru,
    float* __restrict__ gate) {
  __shared__ float wa[64], wb[64], bab[2];
  int t = threadIdx.x;
  if (t < 64) {
    float s = 0.f;
    #pragma unroll
    for (int r = 0; r < 4; ++r) s += Wg[r * 64 + t];
    wa[t] = s;
  } else if (t < 128) {
    int d = t - 64; float s = 0.f;
    #pragma unroll
    for (int r = 4; r < 8; ++r) s += Wg[r * 64 + d];
    wb[d] = s;
  } else if (t == 128) bab[0] = bg[0] + bg[1] + bg[2] + bg[3];
  else if (t == 129)   bab[1] = bg[4] + bg[5] + bg[6] + bg[7];
  __syncthreads();
  int idx = blockIdx.x * 256 + t;
  if (idx >= Bv * Hv * Sv) return;
  int s = idx % Sv, h = (idx / Sv) % Hv, b = idx / (Sv * Hv);
  const unsigned short* xp = xt16 + ((size_t)b * Sv + s) * Cv + h * Dv;
  float da = 0.f, db = 0.f;
  #pragma unroll 8
  for (int d = 0; d < 64; ++d) {
    float x = bf2f(xp[d]);
    da += wa[d] * x; db += wb[d] * x;
  }
  float ga = 1.f / (1.f + __expf(-(da + bab[0])));
  float gb = 1.f / (1.f + __expf(-(db + bab[1])));
  gate[idx] = ga * (gb * gru[h] - 1.f) + 2.f;
}

// ---------------- generic WMMA GEMM: Yt[b,s,o] = Xt[b,s,:]·W[o,:] + bias ----
// Block 256 thr = 8 waves (4x2), tile 128(M=s) x 128(N=o), K-step 32,
// double-buffered LDS fed by async global->LDS copies.
__global__ __launch_bounds__(256) void gemm_bf16_kernel(
    const unsigned short* __restrict__ A,  // (batch, M, K) bf16
    const unsigned short* __restrict__ W,  // (N, K) bf16
    const float* __restrict__ bias,        // (N) or null
    float* __restrict__ outF,              // (batch, M, N) f32 or null
    unsigned short* __restrict__ outH,     // (batch, M, N) bf16 or null
    int M, int N, int K, int fuseGelu) {
  __shared__ __align__(16) unsigned short As[2][128 * 40];
  __shared__ __align__(16) unsigned short Bs[2][128 * 40];
  const int tid = threadIdx.x;
  const int w = tid >> 5, lane = tid & 31;
  const int m16 = lane & 15, half = lane >> 4;
  const int wm = w >> 1, wn = w & 1;       // 4 (M) x 2 (N) waves, 32x64 each
  const int n0 = blockIdx.x * 128;
  const int m0 = blockIdx.y * 128;
  const int bz = blockIdx.z;
  const unsigned short* Ab = A + (size_t)bz * M * K;

  f32x8 acc[2][4];
  #pragma unroll
  for (int mi = 0; mi < 2; ++mi)
    #pragma unroll
    for (int ni = 0; ni < 4; ++ni)
      #pragma unroll
      for (int r = 0; r < 8; ++r) acc[mi][ni][r] = 0.f;

  // issue async copies of one 128x32 A tile + 128x32 B tile into buffer `buf`
  auto load_tiles = [&](int k0, int buf) {
    #pragma unroll
    for (int it = 0; it < 2; ++it) {       // 512 16B chunks / 256 threads
      int c = tid + it * 256;
      int row = c >> 2, part = c & 3;      // 4 chunks (8 bf16 each) per row
      if (m0 + row < M)
        async_copy16(Ab + (size_t)(m0 + row) * K + k0 + part * 8,
                     &As[buf][row * 40 + part * 8]);
    }
    #pragma unroll
    for (int it = 0; it < 2; ++it) {
      int c = tid + it * 256;
      int row = c >> 2, part = c & 3;
      async_copy16(W + (size_t)(n0 + row) * K + k0 + part * 8,
                   &Bs[buf][row * 40 + part * 8]);
    }
  };

  const int nk = K / 32;
  load_tiles(0, 0);
  wait_async();
  __syncthreads();

  #pragma unroll 1
  for (int kt = 0; kt < nk; ++kt) {
    const int buf = kt & 1;
    if (kt + 1 < nk) load_tiles((kt + 1) * 32, buf ^ 1);  // prefetch next tile

    Frag af[2], bfr[4];
    #pragma unroll
    for (int mi = 0; mi < 2; ++mi)
      #pragma unroll
      for (int j = 0; j < 8; ++j)
        af[mi].u[j] =
            *(const unsigned*)&As[buf][(wm * 32 + mi * 16 + m16) * 40 + kA(j, half)];
    #pragma unroll
    for (int ni = 0; ni < 4; ++ni)
      #pragma unroll
      for (int j = 0; j < 8; ++j)
        bfr[ni].u[j] =
            *(const unsigned*)&Bs[buf][(wn * 64 + ni * 16 + m16) * 40 + kB(j, half)];
    #pragma unroll
    for (int mi = 0; mi < 2; ++mi)
      #pragma unroll
      for (int ni = 0; ni < 4; ++ni)
        acc[mi][ni] = wmma_bf16(af[mi], bfr[ni], acc[mi][ni]);

    wait_async();
    __syncthreads();
  }

  #pragma unroll
  for (int mi = 0; mi < 2; ++mi) {
    #pragma unroll
    for (int ni = 0; ni < 4; ++ni) {
      const int o = n0 + wn * 64 + ni * 16 + m16;
      const float bvl = bias ? bias[o] : 0.f;
      #pragma unroll
      for (int r = 0; r < 8; ++r) {
        int s = m0 + wm * 32 + mi * 16 + r + 8 * half;
        if (s >= M) continue;
        float v = acc[mi][ni][r] + bvl;
        if (fuseGelu) v = 0.5f * v * (1.f + erff(v * 0.70710678118f));  // exact GELU
        size_t oi = ((size_t)bz * M + s) * N + o;
        if (outF) outF[oi] = v;
        if (outH) outH[oi] = f2bf(v);
      }
    }
  }
}

// ---------------- flash attention ----------------
// grid (ceil(S/128), H, B); 8 waves/block, each wave owns 16 queries.
__global__ __launch_bounds__(256) void attn_flash_kernel(
    const unsigned short* __restrict__ q16, const unsigned short* __restrict__ k16,
    const unsigned short* __restrict__ v16, const float* __restrict__ gate,
    const float* __restrict__ pb, const float* __restrict__ amask,
    unsigned short* __restrict__ ctx16) {
  __shared__ __align__(16) unsigned short Kt[64 * 72];  // [token][d]
  __shared__ __align__(16) unsigned short Vt[64 * 72];  // [d][token]
  __shared__ __align__(16) unsigned short Pt[8][16 * 72];  // per-wave probs [q][token]

  const int tid = threadIdx.x;
  const int w = tid >> 5, lane = tid & 31;
  const int m16 = lane & 15, half = lane >> 4;
  const int h = blockIdx.y, b = blockIdx.z;
  const int q0 = blockIdx.x * 128 + w * 16;

  Frag qa[2];                                // Q A-frags over d (2 k-steps)
  {
    int qr = q0 + m16; if (qr >= Sv) qr = Sv - 1;
    const unsigned short* qp = q16 + ((size_t)b * Sv + qr) * Cv + h * Dv;
    #pragma unroll
    for (int ks = 0; ks < 2; ++ks)
      #pragma unroll
      for (int j = 0; j < 8; ++j)
        qa[ks].u[j] = *(const unsigned*)(qp + ks * 32 + kA(j, half));
  }
  float gq[8];
  #pragma unroll
  for (int r = 0; r < 8; ++r) {
    int q = q0 + r + 8 * half;
    gq[r] = (q < Sv) ? gate[((size_t)b * Hv + h) * Sv + q] : 0.f;
  }

  float mrow[8], lrow[8];
  f32x8 cacc[4];
  #pragma unroll
  for (int r = 0; r < 8; ++r) { mrow[r] = -3.0e38f; lrow[r] = 0.f; }
  #pragma unroll
  for (int dc = 0; dc < 4; ++dc)
    #pragma unroll
    for (int r = 0; r < 8; ++r) cacc[dc][r] = 0.f;

  const unsigned short* kbase = k16 + (size_t)b * Sv * Cv + h * Dv;
  const unsigned short* vbase = v16 + (size_t)b * Sv * Cv + h * Dv;
  const float* pbh = pb + ((size_t)b * Hv + h) * Sv * Sv;

  const int numKT = (Sv + 63) / 64;
  #pragma unroll 1
  for (int kt = 0; kt < numKT; ++kt) {
    const int k0 = kt * 64;
    __syncthreads();
    // K tile: async global->LDS, 512 16B chunks (stale data in OOB cols is
    // masked to -1e30 below, never accumulated).
    #pragma unroll
    for (int it = 0; it < 2; ++it) {
      int c = tid + it * 256;
      int tok = c >> 3, part = c & 7;
      if (k0 + tok < Sv)
        async_copy16(kbase + (size_t)(k0 + tok) * Cv + part * 8,
                     &Kt[tok * 72 + part * 8]);
    }
    // V tile: manual transposed store (dword reads, 2 ushort LDS writes)
    for (int i = tid; i < 2048; i += 256) {
      int tok = i >> 5, dp = i & 31;
      unsigned val = 0;
      if (k0 + tok < Sv)
        val = *(const unsigned*)(vbase + (size_t)(k0 + tok) * Cv + dp * 2);
      Vt[(dp * 2 + 0) * 72 + tok] = (unsigned short)(val & 0xFFFFu);
      Vt[(dp * 2 + 1) * 72 + tok] = (unsigned short)(val >> 16);
    }
    // prefetch next tile's position_bias lines (the only HBM-bound stream)
    {
      int nq = q0 + (lane & 15), kk2 = k0 + 64 + (lane >> 4) * 32;
      if (nq < Sv && kk2 < Sv)
        __builtin_prefetch(&pbh[(size_t)nq * Sv + kk2], 0, 1);
    }
    wait_async();
    __syncthreads();

    f32x8 sacc[4];                            // scores 16q x 64k
    #pragma unroll
    for (int nc = 0; nc < 4; ++nc)
      #pragma unroll
      for (int r = 0; r < 8; ++r) sacc[nc][r] = 0.f;
    #pragma unroll
    for (int ks = 0; ks < 2; ++ks) {
      #pragma unroll
      for (int nc = 0; nc < 4; ++nc) {
        Frag bk;
        #pragma unroll
        for (int j = 0; j < 8; ++j)
          bk.u[j] = *(const unsigned*)&Kt[(nc * 16 + m16) * 72 + ks * 32 + kB(j, half)];
        sacc[nc] = wmma_bf16(qa[ks], bk, sacc[nc]);
      }
    }

    float tmax[8];
    #pragma unroll
    for (int r = 0; r < 8; ++r) tmax[r] = -3.0e38f;
    #pragma unroll
    for (int nc = 0; nc < 4; ++nc) {
      const int kk = k0 + nc * 16 + m16;
      float kpmv = (kk < Sv) ? (1.f - amask[b * Sv + kk]) * -10000.f : 0.f;
      #pragma unroll
      for (int r = 0; r < 8; ++r) {
        int q = q0 + r + 8 * half;
        float sv = -1.0e30f;
        if (kk < Sv) {
          sv = sacc[nc][r] * 0.125f + kpmv;   // scale = 1/sqrt(64)
          if (q < Sv) sv += gq[r] * pbh[(size_t)q * Sv + kk];
        }
        sacc[nc][r] = sv;
        tmax[r] = fmaxf(tmax[r], sv);
      }
    }
    #pragma unroll
    for (int off = 1; off < 16; off <<= 1)
      #pragma unroll
      for (int r = 0; r < 8; ++r)
        tmax[r] = fmaxf(tmax[r], __shfl_xor(tmax[r], off, 32));

    float rsum[8];
    #pragma unroll
    for (int r = 0; r < 8; ++r) {
      float mnew = fmaxf(mrow[r], tmax[r]);
      float alpha = __expf(mrow[r] - mnew);
      mrow[r] = mnew;
      lrow[r] *= alpha;
      #pragma unroll
      for (int dc = 0; dc < 4; ++dc) cacc[dc][r] *= alpha;
      float s = 0.f;
      #pragma unroll
      for (int nc = 0; nc < 4; ++nc) {
        float p = __expf(sacc[nc][r] - mnew);
        sacc[nc][r] = p;
        s += p;
      }
      rsum[r] = s;
    }
    #pragma unroll
    for (int off = 1; off < 16; off <<= 1)
      #pragma unroll
      for (int r = 0; r < 8; ++r) rsum[r] += __shfl_xor(rsum[r], off, 32);
    #pragma unroll
    for (int r = 0; r < 8; ++r) lrow[r] += rsum[r];

    // probs -> bf16, relayout via private LDS buffer (D-layout -> A-layout)
    unsigned short* myP = &Pt[w][0];
    #pragma unroll
    for (int nc = 0; nc < 4; ++nc)
      #pragma unroll
      for (int r = 0; r < 8; ++r)
        myP[(r + 8 * half) * 72 + nc * 16 + m16] = f2bf(sacc[nc][r]);
    __builtin_amdgcn_wave_barrier();          // same-wave DS in-order; block reorder

    #pragma unroll
    for (int ks = 0; ks < 2; ++ks) {
      Frag pa;
      #pragma unroll
      for (int j = 0; j < 8; ++j)
        pa.u[j] = *(const unsigned*)&myP[m16 * 72 + ks * 32 + kA(j, half)];
      #pragma unroll
      for (int dc = 0; dc < 4; ++dc) {
        Frag bvf;
        #pragma unroll
        for (int j = 0; j < 8; ++j)
          bvf.u[j] = *(const unsigned*)&Vt[(dc * 16 + m16) * 72 + ks * 32 + kB(j, half)];
        cacc[dc] = wmma_bf16(pa, bvf, cacc[dc]);
      }
    }
  }

  #pragma unroll
  for (int r = 0; r < 8; ++r) {
    int q = q0 + r + 8 * half;
    if (q >= Sv) continue;
    float inv = 1.f / lrow[r];
    #pragma unroll
    for (int dc = 0; dc < 4; ++dc)
      ctx16[((size_t)b * Sv + q) * Cv + h * Dv + dc * 16 + m16] =
          f2bf(cacc[dc][r] * inv);
  }
}

// ---------------- LayerNorm over channels, per (b,s) ----------------
__global__ __launch_bounds__(256) void ln_kernel(
    const float* __restrict__ resTokA, const float* __restrict__ resTokB,
    const float* __restrict__ xCmaj, const float* __restrict__ gam,
    const float* __restrict__ bet, float* __restrict__ outTokF,
    unsigned short* __restrict__ outTokH, float* __restrict__ outCmaj) {
  __shared__ float sbuf[8];
  const int idx = blockIdx.x;
  const int b = idx / Sv, s = idx % Sv;
  const int t = threadIdx.x;
  float vals[4];
  float lsum = 0.f;
  #pragma unroll
  for (int i = 0; i < 4; ++i) {
    int c = t + i * 256;
    float v = 0.f;
    if (resTokA) v += resTokA[((size_t)b * Sv + s) * Cv + c];
    if (resTokB) v += resTokB[((size_t)b * Sv + s) * Cv + c];
    if (xCmaj)   v += xCmaj[((size_t)b * Cv + c) * Sv + s];
    vals[i] = v;
    lsum += v;
  }
  float mean = block_sum256(lsum, sbuf) * (1.f / Cv);
  float vsum = 0.f;
  #pragma unroll
  for (int i = 0; i < 4; ++i) { float d = vals[i] - mean; vsum += d * d; }
  float var = block_sum256(vsum, sbuf) * (1.f / Cv);
  float rstd = rsqrtf(var + 1e-5f);
  #pragma unroll
  for (int i = 0; i < 4; ++i) {
    int c = t + i * 256;
    float o = (vals[i] - mean) * rstd * gam[c] + bet[c];
    size_t ti = ((size_t)b * Sv + s) * Cv + c;
    if (outTokF) outTokF[ti] = o;
    if (outTokH) outTokH[ti] = f2bf(o);
    if (outCmaj) outCmaj[((size_t)b * Cv + c) * Sv + s] = o;
  }
}

// ---------------- host orchestration ----------------
extern "C" void kernel_launch(void* const* d_in, const int* in_sizes, int n_in,
                              void* d_out, int out_size, void* d_ws, size_t ws_size,
                              hipStream_t stream) {
  (void)in_sizes; (void)n_in; (void)out_size; (void)ws_size;
  const float* x     = (const float*)d_in[0];
  const float* amask = (const float*)d_in[1];
  const float* pb    = (const float*)d_in[2];
  const float* Wq = (const float*)d_in[3];  const float* bq = (const float*)d_in[4];
  const float* Wk = (const float*)d_in[5];  const float* bk = (const float*)d_in[6];
  const float* Wv = (const float*)d_in[7];  const float* bv = (const float*)d_in[8];
  const float* Wo = (const float*)d_in[9];  const float* bo = (const float*)d_in[10];
  const float* Wg = (const float*)d_in[11]; const float* bg = (const float*)d_in[12];
  const float* gru = (const float*)d_in[13];
  const float* ln1g = (const float*)d_in[14]; const float* ln1b = (const float*)d_in[15];
  const float* W1 = (const float*)d_in[16]; const float* b1 = (const float*)d_in[17];
  const float* W2 = (const float*)d_in[18]; const float* b2 = (const float*)d_in[19];
  const float* ln2g = (const float*)d_in[20]; const float* ln2b = (const float*)d_in[21];
  float* out = (float*)d_out;

  const size_t CC  = (size_t)Cv * Cv;       // 1,048,576
  const size_t FC  = (size_t)Fv * Cv;       // 4,194,304
  const size_t BSC = (size_t)Bv * Sv * Cv;  // 3,072,000
  char* ws = (char*)d_ws;
  size_t off = 0;
  auto take = [&](size_t bytes) { size_t o = off; off += (bytes + 255) & ~(size_t)255; return o; };
  size_t oWQ = take(CC * 2), oWK = take(CC * 2), oWV = take(CC * 2), oWO = take(CC * 2);
  size_t oW1 = take(FC * 2), oW2 = take(FC * 2);
  size_t oXT = take(BSC * 2);                       // later reused as H16
  size_t oGT = take((size_t)Bv * Hv * Sv * 4);
  size_t oUN = take(BSC * 2 * 4);                   // q/k/v/ctx OR F16 (24.58 MB)
  size_t oAO = take(BSC * 4);                       // attn_out f32, reused as Y32
  size_t oH32 = take(BSC * 4);

  unsigned short* WQ16 = (unsigned short*)(ws + oWQ);
  unsigned short* WK16 = (unsigned short*)(ws + oWK);
  unsigned short* WV16 = (unsigned short*)(ws + oWV);
  unsigned short* WO16 = (unsigned short*)(ws + oWO);
  unsigned short* W116 = (unsigned short*)(ws + oW1);
  unsigned short* W216 = (unsigned short*)(ws + oW2);
  unsigned short* XT16 = (unsigned short*)(ws + oXT);
  unsigned short* H16  = (unsigned short*)(ws + oXT);   // alias (XT dead post-QKV)
  float*          GATE = (float*)(ws + oGT);
  unsigned short* Q16  = (unsigned short*)(ws + oUN);
  unsigned short* K16  = (unsigned short*)(ws + oUN + BSC * 2);
  unsigned short* V16  = (unsigned short*)(ws + oUN + BSC * 4);
  unsigned short* CTX16= (unsigned short*)(ws + oUN + BSC * 6);
  unsigned short* F16  = (unsigned short*)(ws + oUN);   // alias (q/k/v/ctx dead)
  float*          AO32 = (float*)(ws + oAO);
  float*          Y32  = (float*)(ws + oAO);            // alias (AO dead post-LN1)
  float*          H32  = (float*)(ws + oH32);

  const int mTiles = (Sv + 127) / 128;    // 12
  const int qBlks  = (Sv + 127) / 128;    // 12

  // 1. weight + activation converts (bf16)
  cvt_f32_bf16_kernel<<<(int)((CC + 255) / 256), 256, 0, stream>>>(Wq, WQ16, (int)CC);
  cvt_f32_bf16_kernel<<<(int)((CC + 255) / 256), 256, 0, stream>>>(Wk, WK16, (int)CC);
  cvt_f32_bf16_kernel<<<(int)((CC + 255) / 256), 256, 0, stream>>>(Wv, WV16, (int)CC);
  cvt_f32_bf16_kernel<<<(int)((CC + 255) / 256), 256, 0, stream>>>(Wo, WO16, (int)CC);
  cvt_f32_bf16_kernel<<<(int)((FC + 255) / 256), 256, 0, stream>>>(W1, W116, (int)FC);
  cvt_f32_bf16_kernel<<<(int)((FC + 255) / 256), 256, 0, stream>>>(W2, W216, (int)FC);
  xpose_kernel<<<(int)((BSC + 255) / 256), 256, 0, stream>>>(x, XT16);

  // 2. gate
  gate_kernel<<<(Bv * Hv * Sv + 255) / 256, 256, 0, stream>>>(XT16, Wg, bg, gru, GATE);

  // 3. QKV projections
  gemm_bf16_kernel<<<dim3(Cv / 128, mTiles, Bv), 256, 0, stream>>>(
      XT16, WQ16, bq, nullptr, Q16, Sv, Cv, Cv, 0);
  gemm_bf16_kernel<<<dim3(Cv / 128, mTiles, Bv), 256, 0, stream>>>(
      XT16, WK16, bk, nullptr, K16, Sv, Cv, Cv, 0);
  gemm_bf16_kernel<<<dim3(Cv / 128, mTiles, Bv), 256, 0, stream>>>(
      XT16, WV16, bv, nullptr, V16, Sv, Cv, Cv, 0);

  // 4. flash attention (position_bias streamed once)
  attn_flash_kernel<<<dim3(qBlks, Hv, Bv), 256, 0, stream>>>(
      Q16, K16, V16, GATE, pb, amask, CTX16);

  // 5. output projection + LN1(h = LN(x + attn_out))
  gemm_bf16_kernel<<<dim3(Cv / 128, mTiles, Bv), 256, 0, stream>>>(
      CTX16, WO16, bo, AO32, nullptr, Sv, Cv, Cv, 0);
  ln_kernel<<<Bv * Sv, 256, 0, stream>>>(nullptr, AO32, x, ln1g, ln1b,
                                         H32, H16, nullptr);

  // 6. FFN + LN2 -> out (B,C,1,S)
  gemm_bf16_kernel<<<dim3(Fv / 128, mTiles, Bv), 256, 0, stream>>>(
      H16, W116, b1, nullptr, F16, Sv, Fv, Cv, 1 /*gelu*/);
  gemm_bf16_kernel<<<dim3(Cv / 128, mTiles, Bv), 256, 0, stream>>>(
      F16, W216, b2, Y32, nullptr, Sv, Cv, Fv, 0);
  ln_kernel<<<Bv * Sv, 256, 0, stream>>>(H32, Y32, nullptr, ln2g, ln2b,
                                         nullptr, nullptr, out);
}
